// S2S_LSTM_75076028334780
// MI455X (gfx1250) — matrix operand order repaired and checked by reading, not compiled, not run on hardware
//
#include <hip/hip_runtime.h>
#include <math.h>

// Problem constants (fixed by the reference)
#define BB  64    // batch
#define SS  256   // encoder sequence length
#define CC  64    // input / output channels
#define HH  512   // hidden size
#define TT  128   // decoder steps
#define G4H 2048  // 4*H (gate dimension)

typedef __attribute__((ext_vector_type(16))) __bf16 v16bf;
typedef __attribute__((ext_vector_type(8)))  __bf16 v8bf;
typedef __attribute__((ext_vector_type(8)))  float  v8f;
typedef unsigned short u16;

// ---------- helpers ----------

// fp32 -> bf16 bits, round-to-nearest-even
__device__ __forceinline__ u16 f2bf(float x) {
  unsigned u = __builtin_bit_cast(unsigned, x);
  u += 0x7FFFu + ((u >> 16) & 1u);
  return (u16)(u >> 16);
}

__device__ __forceinline__ float sigf(float x) {
  return 1.0f / (1.0f + __expf(-x));
}

// A fragment, 16x32 bf16 (contiguous k order, not the ISA table order): lane l = row (l&15);
// K = kh..kh+15 contiguously, kh=(l>>4)*16.
__device__ __forceinline__ v16bf mk_fragA(const u16* p) {
  v8bf lo = *(const v8bf*)(p);       // 16B
  v8bf hi = *(const v8bf*)(p + 8);   // 16B
  return __builtin_shufflevector(lo, hi,
      0, 1, 2, 3, 4, 5, 6, 7, 8, 9, 10, 11, 12, 13, 14, 15);
}

// B fragment, 32x16 bf16 (contiguous k order, same as A): lane l = column (l&15);
// lanes 0-15 hold K=0..15 contiguously, lanes 16-31 hold K=16..31.
__device__ __forceinline__ v16bf mk_fragB(const u16* p) {
  v8bf lo = *(const v8bf*)(p);       // 16B
  v8bf hi = *(const v8bf*)(p + 8);   // 16B
  return __builtin_shufflevector(lo, hi,
      0, 1, 2, 3, 4, 5, 6, 7, 8, 9, 10, 11, 12, 13, 14, 15);
}

__device__ __forceinline__ v8f wmma_bf16(v16bf a, v16bf b, v8f c) {
  return __builtin_amdgcn_wmma_f32_16x16x32_bf16(
      false, a, false, b, (short)0, c, false, false);
}

// 4-gate GEMM slice. All fragment loads for one K-block are grouped before
// the 4 WMMAs (one load-wait per iteration); no software pipeline -- with
// 8 waves/SIMD resident, cross-wave parallelism hides the L2 latency and
// the register footprint stays within the 32-wave/WGP budget (no spills).
// K and GS (gate stride, elements) are compile-time so the 3 gate offsets
// fold into the 24-bit load IOFFSET -- a single address register.
template <int K, int GS>
__device__ __forceinline__ void gemm4(const u16* __restrict__ ap,
                                      const u16* __restrict__ wp,
                                      v8f& a0, v8f& a1, v8f& a2, v8f& a3) {
#pragma unroll 1
  for (int kk = 0; kk < K; kk += 32) {
    v16bf af = mk_fragA(ap + kk);
    v16bf b0 = mk_fragB(wp + kk);
    v16bf b1 = mk_fragB(wp + kk + GS);
    v16bf b2 = mk_fragB(wp + kk + 2 * GS);
    v16bf b3 = mk_fragB(wp + kk + 3 * GS);
    a0 = wmma_bf16(af, b0, a0);
    a1 = wmma_bf16(af, b1, a1);
    a2 = wmma_bf16(af, b2, a2);
    a3 = wmma_bf16(af, b3, a3);
  }
}

// single-tile GEMM (output linear)
template <int K>
__device__ __forceinline__ void gemm1(const u16* __restrict__ ap,
                                      const u16* __restrict__ wp, v8f& a0) {
#pragma unroll 1
  for (int kk = 0; kk < K; kk += 32) {
    v16bf af = mk_fragA(ap + kk);
    v16bf b0 = mk_fragB(wp + kk);
    a0 = wmma_bf16(af, b0, a0);
  }
}

// ---------- fp32 -> bf16 packing ----------

__global__ void pack_bf16_kernel(const float* __restrict__ src,
                                 u16* __restrict__ dst, int n) {
  int i = blockIdx.x * blockDim.x + threadIdx.x;
  if (i < n) dst[i] = f2bf(src[i]);
}

// ---------- persistent single-workgroup LSTM layer ----------
//
// 1024 threads = 32 waves. Wave w: mt = w&3 (16-row batch tile),
// ng = w>>2 (64-column group). 4 unrolled passes t=0..3; pass t owns
// columns j = ng*64 + t*16 + (l&15) of all four gates (4 live accumulators).
// h is double-buffered in LDS (ping-pong) -> pointwise writes h_next with no
// barrier before it; 1 barrier per time step. c persists in registers.
template <int KIN>
__global__ void __launch_bounds__(1024, 1)
lstm_layer_kernel(const u16* __restrict__ xin,     // (B, steps, KIN) bf16
                  int xrow,                        // row stride = steps*KIN
                  const u16* __restrict__ Wih,     // (4H, KIN) bf16 row-major
                  const u16* __restrict__ Whh,     // (4H, H)  bf16 row-major
                  const float* __restrict__ bias,  // (4H,) f32
                  u16* __restrict__ yout,          // (B, steps, H) or null
                  u16* __restrict__ hT,            // (B, H) bf16 final hidden
                  float* __restrict__ cT,          // (B, H) f32 final cell
                  int steps) {
  extern __shared__ u16 smem[];
  u16* hbuf0 = smem;            // 64x512 bf16
  u16* hbuf1 = smem + BB * HH;  // 64x512 bf16

  const int tid  = threadIdx.x;
  const int lane = tid & 31;
  const int wid  = tid >> 5;
  const int mt   = wid & 3;
  const int ng   = wid >> 2;
  const int l15  = lane & 15;
  const int kha  = (lane >> 4) * 16;  // A-fragment K offset
  const int khb  = (lane >> 4) * 16;  // B-fragment K offset
  const int rowA = mt * 16 + l15;
  const int mrow = mt * 16 + (lane >> 4) * 8;

  for (int i = tid; i < BB * HH; i += 1024) hbuf0[i] = 0;

  float creg[4][8];
#pragma unroll
  for (int t = 0; t < 4; ++t)
#pragma unroll
    for (int r = 0; r < 8; ++r) creg[t][r] = 0.0f;

  const v8f vzero = {0.f, 0.f, 0.f, 0.f, 0.f, 0.f, 0.f, 0.f};

  __syncthreads();

  for (int st = 0; st < steps; ++st) {
    u16* hcur = (st & 1) ? hbuf1 : hbuf0;
    u16* hnxt = (st & 1) ? hbuf0 : hbuf1;
    const u16* xbase = xin + st * KIN + rowA * xrow + kha;

#pragma unroll
    for (int t = 0; t < 4; ++t) {
      const int j = ng * 64 + t * 16 + l15;
      v8f a0 = vzero, a1 = vzero, a2 = vzero, a3 = vzero;

      // gates += x_t @ Wih^T
      gemm4<KIN, HH * KIN>(xbase, Wih + j * KIN + khb, a0, a1, a2, a3);
      // gates += h_{t-1} @ Whh^T (A from LDS)
      gemm4<HH, HH * HH>(hcur + rowA * HH + kha, Whh + j * HH + khb,
                         a0, a1, a2, a3);

      // LSTM pointwise; write h_t straight into the other LDS buffer
      const float b0 = bias[j];
      const float b1 = bias[HH + j];
      const float b2 = bias[2 * HH + j];
      const float b3 = bias[3 * HH + j];
#pragma unroll
      for (int r = 0; r < 8; ++r) {
        const float iv = sigf(a0[r] + b0);
        const float fv = sigf(a1[r] + b1);
        const float gv = tanhf(a2[r] + b2);
        const float ov = sigf(a3[r] + b3);
        const float c  = fv * creg[t][r] + iv * gv;
        creg[t][r] = c;
        const u16 hb16 = f2bf(ov * tanhf(c));
        hnxt[(mrow + r) * HH + j] = hb16;
        if (yout) yout[((mrow + r) * steps + st) * HH + j] = hb16;
      }
    }
    __syncthreads();  // h_t visible; next step flips buffers
  }

  // final states for the decoder
  const u16* hfin = (steps & 1) ? hbuf1 : hbuf0;
#pragma unroll
  for (int t = 0; t < 4; ++t) {
    const int j = ng * 64 + t * 16 + l15;
#pragma unroll
    for (int r = 0; r < 8; ++r) {
      const int m = mrow + r;
      cT[m * HH + j] = creg[t][r];
      hT[m * HH + j] = hfin[m * HH + j];
    }
  }
}

// ---------- persistent decoder: 2 LSTM layers + output linear per step ----------
__global__ void __launch_bounds__(1024, 1)
decoder_kernel(const u16* __restrict__ xbf,  // (B,S,C) bf16 (for x[:,-1,:])
               const u16* __restrict__ h0i, const float* __restrict__ c0i,
               const u16* __restrict__ h1i, const float* __restrict__ c1i,
               const u16* __restrict__ Wih0, const u16* __restrict__ Whh0,
               const float* __restrict__ b0v,
               const u16* __restrict__ Wih1, const u16* __restrict__ Whh1,
               const float* __restrict__ b1v,
               const u16* __restrict__ linW,  // (C,H) bf16
               const float* __restrict__ linb,
               float* __restrict__ out) {  // (B,T,C) f32
  extern __shared__ u16 smem[];
  u16* h0b0 = smem;                  // 64x512
  u16* h0b1 = smem + BB * HH;        // 64x512
  u16* h1b0 = smem + 2 * BB * HH;    // 64x512
  u16* h1b1 = smem + 3 * BB * HH;    // 64x512
  u16* in_lds = smem + 4 * BB * HH;  // 64x64 feedback input

  const int tid  = threadIdx.x;
  const int lane = tid & 31;
  const int wid  = tid >> 5;
  const int mt   = wid & 3;
  const int ng   = wid >> 2;
  const int l15  = lane & 15;
  const int kha  = (lane >> 4) * 16;
  const int khb  = (lane >> 4) * 16;
  const int rowA = mt * 16 + l15;
  const int mrow = mt * 16 + (lane >> 4) * 8;

  for (int i = tid; i < BB * HH; i += 1024) {
    h0b0[i] = h0i[i];
    h1b0[i] = h1i[i];
  }
  for (int i = tid; i < BB * CC; i += 1024) {
    const int m = i / CC, j = i % CC;
    in_lds[i] = xbf[(m * SS + (SS - 1)) * CC + j];  // x[:, -1, :]
  }

  float c0reg[4][8], c1reg[4][8];
#pragma unroll
  for (int t = 0; t < 4; ++t) {
    const int j = ng * 64 + t * 16 + l15;
#pragma unroll
    for (int r = 0; r < 8; ++r) {
      c0reg[t][r] = c0i[(mrow + r) * HH + j];
      c1reg[t][r] = c1i[(mrow + r) * HH + j];
    }
  }

  const v8f vzero = {0.f, 0.f, 0.f, 0.f, 0.f, 0.f, 0.f, 0.f};
  __syncthreads();

  for (int st = 0; st < TT; ++st) {
    u16* h0c = (st & 1) ? h0b1 : h0b0;
    u16* h0n = (st & 1) ? h0b0 : h0b1;
    u16* h1c = (st & 1) ? h1b1 : h1b0;
    u16* h1n = (st & 1) ? h1b0 : h1b1;

    // ================= layer 0: in (C=64) + h0 (H) =================
#pragma unroll
    for (int t = 0; t < 4; ++t) {
      const int j = ng * 64 + t * 16 + l15;
      v8f a0 = vzero, a1 = vzero, a2 = vzero, a3 = vzero;
      gemm4<CC, HH * CC>(in_lds + rowA * CC + kha, Wih0 + j * CC + khb,
                         a0, a1, a2, a3);
      gemm4<HH, HH * HH>(h0c + rowA * HH + kha, Whh0 + j * HH + khb,
                         a0, a1, a2, a3);
      const float g0 = b0v[j];
      const float g1 = b0v[HH + j];
      const float g2 = b0v[2 * HH + j];
      const float g3 = b0v[3 * HH + j];
#pragma unroll
      for (int r = 0; r < 8; ++r) {
        const float iv = sigf(a0[r] + g0);
        const float fv = sigf(a1[r] + g1);
        const float gv = tanhf(a2[r] + g2);
        const float ov = sigf(a3[r] + g3);
        const float c  = fv * c0reg[t][r] + iv * gv;
        c0reg[t][r] = c;
        h0n[(mrow + r) * HH + j] = f2bf(ov * tanhf(c));
      }
    }
    __syncthreads();  // h0_t visible

    // ================= layer 1: h0new (H) + h1 (H) =================
#pragma unroll
    for (int t = 0; t < 4; ++t) {
      const int j = ng * 64 + t * 16 + l15;
      v8f a0 = vzero, a1 = vzero, a2 = vzero, a3 = vzero;
      gemm4<HH, HH * HH>(h0n + rowA * HH + kha, Wih1 + j * HH + khb,
                         a0, a1, a2, a3);
      gemm4<HH, HH * HH>(h1c + rowA * HH + kha, Whh1 + j * HH + khb,
                         a0, a1, a2, a3);
      const float g0 = b1v[j];
      const float g1 = b1v[HH + j];
      const float g2 = b1v[2 * HH + j];
      const float g3 = b1v[3 * HH + j];
#pragma unroll
      for (int r = 0; r < 8; ++r) {
        const float iv = sigf(a0[r] + g0);
        const float fv = sigf(a1[r] + g1);
        const float gv = tanhf(a2[r] + g2);
        const float ov = sigf(a3[r] + g3);
        const float c  = fv * c1reg[t][r] + iv * gv;
        c1reg[t][r] = c;
        h1n[(mrow + r) * HH + j] = f2bf(ov * tanhf(c));
      }
    }
    __syncthreads();  // h1_t visible

    // ========== output linear: out = h1 @ linW^T + linb (waves 0..15) ====
    if (wid < 16) {                 // tiles: mt in 0..3, ng in 0..3
      const int j = ng * 16 + l15;  // output channel 0..63
      v8f ao = vzero;
      gemm1<HH>(h1n + rowA * HH + kha, linW + j * HH + khb, ao);
      const float lb = linb[j];
#pragma unroll
      for (int r = 0; r < 8; ++r) {
        const int m = mrow + r;
        const float o = ao[r] + lb;
        out[(m * TT + st) * CC + j] = o;  // (B,T,C)
        in_lds[m * CC + j] = f2bf(o);     // feedback for next step
      }
    }
    __syncthreads();  // feedback input visible for next step
  }
}

// ---------- host-side launch ----------

extern "C" void kernel_launch(void* const* d_in, const int* in_sizes, int n_in,
                              void* d_out, int out_size, void* d_ws,
                              size_t ws_size, hipStream_t stream) {
  (void)in_sizes; (void)n_in; (void)out_size; (void)ws_size;
  const float* x     = (const float*)d_in[0];
  const float* eWih0 = (const float*)d_in[1];
  const float* eWhh0 = (const float*)d_in[2];
  const float* eb0   = (const float*)d_in[3];
  const float* eWih1 = (const float*)d_in[4];
  const float* eWhh1 = (const float*)d_in[5];
  const float* eb1   = (const float*)d_in[6];
  const float* dWih0 = (const float*)d_in[7];
  const float* dWhh0 = (const float*)d_in[8];
  const float* db0   = (const float*)d_in[9];
  const float* dWih1 = (const float*)d_in[10];
  const float* dWhh1 = (const float*)d_in[11];
  const float* db1   = (const float*)d_in[12];
  const float* linW  = (const float*)d_in[13];
  const float* linb  = (const float*)d_in[14];

  // Workspace layout (bf16 region, then f32 region). ~34 MB total.
  u16* w = (u16*)d_ws;
  size_t off = 0;
  auto take = [&](size_t n) { u16* p = w + off; off += n; return p; };
  u16* xb     = take((size_t)BB * SS * CC);
  u16* eWih0b = take((size_t)G4H * CC);
  u16* eWhh0b = take((size_t)G4H * HH);
  u16* eWih1b = take((size_t)G4H * HH);
  u16* eWhh1b = take((size_t)G4H * HH);
  u16* dWih0b = take((size_t)G4H * CC);
  u16* dWhh0b = take((size_t)G4H * HH);
  u16* dWih1b = take((size_t)G4H * HH);
  u16* dWhh1b = take((size_t)G4H * HH);
  u16* linWb  = take((size_t)CC * HH);
  u16* y0b    = take((size_t)BB * SS * HH);
  u16* h0T    = take((size_t)BB * HH);
  u16* h1T    = take((size_t)BB * HH);
  off = (off + 1) & ~(size_t)1;  // 4B-align the f32 region
  float* c0T = (float*)(w + off); off += 2 * (size_t)BB * HH;
  float* c1T = (float*)(w + off); off += 2 * (size_t)BB * HH;

  auto pack = [&](const float* s, u16* d, int n) {
    pack_bf16_kernel<<<(n + 255) / 256, 256, 0, stream>>>(s, d, n);
  };
  pack(x, xb, BB * SS * CC);
  pack(eWih0, eWih0b, G4H * CC);
  pack(eWhh0, eWhh0b, G4H * HH);
  pack(eWih1, eWih1b, G4H * HH);
  pack(eWhh1, eWhh1b, G4H * HH);
  pack(dWih0, dWih0b, G4H * CC);
  pack(dWhh0, dWhh0b, G4H * HH);
  pack(dWih1, dWih1b, G4H * HH);
  pack(dWhh1, dWhh1b, G4H * HH);
  pack(linW, linWb, CC * HH);

  const size_t lds1 = (size_t)(2 * BB * HH) * sizeof(u16);  // 128 KB
  lstm_layer_kernel<CC><<<1, 1024, lds1, stream>>>(
      xb, SS * CC, eWih0b, eWhh0b, eb0, y0b, h0T, c0T, SS);
  lstm_layer_kernel<HH><<<1, 1024, lds1, stream>>>(
      y0b, SS * HH, eWih1b, eWhh1b, eb1, nullptr, h1T, c1T, SS);

  const size_t lds2 = (size_t)(4 * BB * HH + BB * CC) * sizeof(u16);  // 264 KB
  decoder_kernel<<<1, 1024, lds2, stream>>>(
      xb, h0T, c0T, h1T, c1T, dWih0b, dWhh0b, db0, dWih1b, dWhh1b, db1,
      linWb, linb, (float*)d_out);
}
